// MultiHeadAttention_71219147702647
// MI455X (gfx1250) — matrix operand order repaired and checked by reading, not compiled, not run on hardware
//
#include <hip/hip_runtime.h>
#include <hip/hip_bf16.h>
#include <math.h>

// ---------------- problem constants ----------------
#define BN 2
#define TT 2048
#define CC 768
#define HH 12
#define HD 64

typedef _Float16 h16 __attribute__((ext_vector_type(16)));
typedef _Float16 h8v __attribute__((ext_vector_type(8)));
typedef float    f8  __attribute__((ext_vector_type(8)));

#define WMMA_F16(a, b, c) \
  __builtin_amdgcn_wmma_f32_16x16x32_f16(false, (a), false, (b), (short)0, (c), false, false)

// ============================================================
// WMMA fragment loaders — all paths are 16-byte vector loads.
//
// A (16x32, f16): lane L holds row M=L&15, K = {koff..koff+7, 16+koff..+7},
//                 koff = 8*(L>>4).
// B (32x16, f16): lane L holds column N=L&15, K = {kb..kb+7, 16+kb..+7}, kb = 8*(L>>4).
// C/D (16x16,f32): VGPR r: lanes 0-15 -> (M=r, N=lane); lanes 16-31 -> M=r+8.
// ============================================================
static __device__ __forceinline__ h16 cat8(h8v lo, h8v hi) {
  return __builtin_shufflevector(lo, hi, 0, 1, 2, 3, 4, 5, 6, 7,
                                 8, 9, 10, 11, 12, 13, 14, 15);
}

static __device__ __forceinline__ h16 load_a16(const _Float16* base, int ld) {
  const int lane = threadIdx.x & 31;
  const int m    = lane & 15;
  const int koff = (lane >> 4) << 3;
  const _Float16* p = base + (size_t)m * ld + koff;
  return cat8(*(const h8v*)p, *(const h8v*)(p + 16));   // 2x global_load_b128
}

// B[k][n] = mem[n*ld + k] ("nk" memory) -> two 8-half runs per lane (same K as A)
static __device__ __forceinline__ h16 load_b16(const _Float16* base, int ld) {
  const int lane = threadIdx.x & 31;
  const int n    = lane & 15;
  const int kb   = (lane >> 4) << 3;
  const _Float16* p = base + (size_t)n * ld + kb;
  return cat8(*(const h8v*)p, *(const h8v*)(p + 16));   // 2x global_load_b128
}

// ============================================================
// Kernel 0a: streaming fp32 -> f16 conversion (8 elems / thread)
// ============================================================
__global__ void cvt_f16_kernel(const float* __restrict__ in,
                               _Float16* __restrict__ out, int n8) {
  const int i = blockIdx.x * blockDim.x + threadIdx.x;
  if (i >= n8) return;
  const float* p = in + (size_t)i * 8;
  h8v o;
#pragma unroll
  for (int j = 0; j < 8; ++j) o[j] = (_Float16)p[j];
  *((h8v*)out + i) = o;
}

// ============================================================
// Kernel 0b: W[H,C,HD] fp32 -> Wt[H,HD,C] f16 (transpose per head)
// one thread per (h, d, 8 consecutive c)
// ============================================================
__global__ void wqkv_t_kernel(const float* __restrict__ Wq,
                              const float* __restrict__ Wk,
                              const float* __restrict__ Wv,
                              _Float16* __restrict__ Wqt,
                              _Float16* __restrict__ Wkt,
                              _Float16* __restrict__ Wvt) {
  const int z = blockIdx.z;
  const float* W = (z == 0) ? Wq : ((z == 1) ? Wk : Wv);
  _Float16*   Wt = (z == 0) ? Wqt : ((z == 1) ? Wkt : Wvt);

  const int idx = blockIdx.x * blockDim.x + threadIdx.x;  // over H*HD*(C/8)
  if (idx >= HH * HD * (CC / 8)) return;
  const int c8 = idx % (CC / 8);
  const int hd = idx / (CC / 8);
  const int d  = hd % HD;
  const int h  = hd / HD;

  const float* src = W + ((size_t)h * CC + c8 * 8) * HD + d;
  h8v o;
#pragma unroll
  for (int j = 0; j < 8; ++j) o[j] = (_Float16)src[(size_t)j * HD];
  *(h8v*)(Wt + ((size_t)h * HD + d) * CC + c8 * 8) = o;
}

// ============================================================
// Kernel 1: QKV projection GEMM, all-f16 operands, fp32 accum.
// grid = (B*T/32, H, 3), block = 32 (one wave). 32x64 tile/wave.
// Q gets 1/sqrt(HD); V is written TRANSPOSED [B,H,HD,T] with
// vector b128 stores (C-layout rows are contiguous per lane there).
// ============================================================
__global__ void qkv_gemm_kernel(const _Float16* __restrict__ Xq,
                                const _Float16* __restrict__ Xk,
                                const _Float16* __restrict__ Xv,
                                const _Float16* __restrict__ Wqt,
                                const _Float16* __restrict__ Wkt,
                                const _Float16* __restrict__ Wvt,
                                _Float16* __restrict__ Qo,
                                _Float16* __restrict__ Ko,
                                _Float16* __restrict__ VtO) {
  const int z = blockIdx.z;
  const _Float16* X  = (z == 0) ? Xq : ((z == 1) ? Xk : Xv);
  const _Float16* Wt = (z == 0) ? Wqt : ((z == 1) ? Wkt : Wvt);

  const int h  = blockIdx.y;
  const int bt = blockIdx.x;
  const int b  = bt / (TT / 32);
  const int t0 = (bt % (TT / 32)) * 32;

  const _Float16* Xb = X + ((size_t)b * TT + t0) * CC;
  const _Float16* Wh = Wt + (size_t)h * HD * CC;

  f8 acc[2][4] = {};
  for (int k0 = 0; k0 < CC; k0 += 32) {
    if (k0 + 32 < CC) __builtin_prefetch(Xb + k0 + 32, 0, 0);  // global_prefetch_b8
    h16 a0 = load_a16(Xb + k0, CC);
    h16 a1 = load_a16(Xb + (size_t)16 * CC + k0, CC);
#pragma unroll
    for (int t = 0; t < 4; ++t) {
      h16 bm = load_b16(Wh + (size_t)(t * 16) * CC + k0, CC);
      acc[0][t] = WMMA_F16(a0, bm, acc[0][t]);
      acc[1][t] = WMMA_F16(a1, bm, acc[1][t]);
    }
  }

  const int lane = threadIdx.x & 31;
  const int nn   = lane & 15;
  const int hi   = lane >> 4;

  if (z == 2) {
    // V stored transposed: Vt[d][t], contiguous along t per lane.
    _Float16* Vb = VtO + (size_t)(b * HH + h) * HD * TT;
#pragma unroll
    for (int rt = 0; rt < 2; ++rt)
#pragma unroll
      for (int t = 0; t < 4; ++t) {
        h8v o;
#pragma unroll
        for (int r = 0; r < 8; ++r) o[r] = (_Float16)acc[rt][t][r];
        *(h8v*)(Vb + (size_t)(t * 16 + nn) * TT + t0 + rt * 16 + 8 * hi) = o;
      }
  } else {
    const float scale = (z == 0) ? 0.125f : 1.0f;
    _Float16* Ob = ((z == 0) ? Qo : Ko) + ((size_t)(b * HH + h) * TT + t0) * HD;
#pragma unroll
    for (int rt = 0; rt < 2; ++rt)
#pragma unroll
      for (int t = 0; t < 4; ++t)
#pragma unroll
        for (int r = 0; r < 8; ++r) {
          const int M = rt * 16 + r + 8 * hi;
          Ob[(size_t)M * HD + t * 16 + nn] = (_Float16)(acc[rt][t][r] * scale);
        }
  }
}

// ============================================================
// Kernel 2: causal flash attention, 32-row tile per wave.
// grid = (T/32, H, B), block = 32.
// K/V fragments shared across both 16-row subtiles.
// P converted C-layout -> A-layout through 2KB LDS (b128 ds loads).
// ============================================================
__global__ void attn_kernel(const _Float16* __restrict__ Q,
                            const _Float16* __restrict__ K,
                            const _Float16* __restrict__ Vt,
                            _Float16* __restrict__ Out) {
  __shared__ __align__(16) _Float16 Plds[32 * 32];

  const int b  = blockIdx.z;
  const int h  = blockIdx.y;
  const int t0 = blockIdx.x * 32;

  const size_t hoff = (size_t)(b * HH + h) * TT * HD;
  const _Float16* Qh = Q + hoff;
  const _Float16* Kh = K + hoff;
  const _Float16* Vh = Vt + hoff;   // [HD][T] per head

  const int lane = threadIdx.x & 31;
  const int hi   = lane >> 4;
  const int nn   = lane & 15;

  h16 qa[2][2];
#pragma unroll
  for (int rt = 0; rt < 2; ++rt) {
    qa[rt][0] = load_a16(Qh + (size_t)(t0 + rt * 16) * HD, HD);
    qa[rt][1] = load_a16(Qh + (size_t)(t0 + rt * 16) * HD + 32, HD);
  }

  f8 o[2][4] = {};
  float m[2][8], l[2][8];
#pragma unroll
  for (int rt = 0; rt < 2; ++rt)
#pragma unroll
    for (int r = 0; r < 8; ++r) { m[rt][r] = -3.0e38f; l[rt][r] = 0.0f; }

  const int send = t0 + 32;  // exclusive causal col bound
  for (int s0 = 0; s0 < send; s0 += 32) {
    if (s0 + 32 < send) __builtin_prefetch(Kh + (size_t)(s0 + 32) * HD, 0, 0);

    // ---- S = Q.K^T : four 16x16 tiles, K fragments shared ----
    f8 st[2][2];
#pragma unroll
    for (int sb = 0; sb < 2; ++sb) {
      const int sc = s0 + sb * 16;
      h16 kb0 = load_b16(Kh + (size_t)sc * HD, HD);
      h16 kb1 = load_b16(Kh + (size_t)sc * HD + 32, HD);
#pragma unroll
      for (int rt = 0; rt < 2; ++rt) {
        f8 s = {};
        if (sc < t0 + rt * 16 + 16) {  // wave-uniform: skip fully-masked tiles
          s = WMMA_F16(qa[rt][0], kb0, s);
          s = WMMA_F16(qa[rt][1], kb1, s);
        }
#pragma unroll
        for (int r = 0; r < 8; ++r) {
          const int row = t0 + rt * 16 + r + 8 * hi;
          const int col = sc + nn;
          if (col > row) s[r] = -3.0e38f;
        }
        st[rt][sb] = s;
      }
    }

    // ---- online softmax per 16-row subtile ----
#pragma unroll
    for (int rt = 0; rt < 2; ++rt) {
      float fac[8];
#pragma unroll
      for (int r = 0; r < 8; ++r) {
        float v = fmaxf(st[rt][0][r], st[rt][1][r]);
        v = fmaxf(v, __shfl_xor(v, 1, 32));
        v = fmaxf(v, __shfl_xor(v, 2, 32));
        v = fmaxf(v, __shfl_xor(v, 4, 32));
        v = fmaxf(v, __shfl_xor(v, 8, 32));
        const float mnew = fmaxf(m[rt][r], v);
        fac[r] = __expf(m[rt][r] - mnew);
        m[rt][r] = mnew;
      }
#pragma unroll
      for (int sb = 0; sb < 2; ++sb)
#pragma unroll
        for (int r = 0; r < 8; ++r)
          st[rt][sb][r] = __expf(st[rt][sb][r] - m[rt][r]);
#pragma unroll
      for (int r = 0; r < 8; ++r) {
        float rs = st[rt][0][r] + st[rt][1][r];
        rs += __shfl_xor(rs, 1, 32);
        rs += __shfl_xor(rs, 2, 32);
        rs += __shfl_xor(rs, 4, 32);
        rs += __shfl_xor(rs, 8, 32);
        l[rt][r] = l[rt][r] * fac[r] + rs;
#pragma unroll
        for (int t = 0; t < 4; ++t) o[rt][t][r] *= fac[r];
        const int M = rt * 16 + r + 8 * hi;
        Plds[M * 32 + nn]      = (_Float16)st[rt][0][r];   // ds_store_b16
        Plds[M * 32 + 16 + nn] = (_Float16)st[rt][1][r];
      }
    }
    __syncthreads();  // single-wave WG -> LDS waits

    // ---- reload P as A fragments (b128 ds loads) ----
    h16 pa[2];
    {
      const int mrow = lane & 15;
      const int koff = hi * 8;
#pragma unroll
      for (int rt = 0; rt < 2; ++rt) {
        const _Float16* p = Plds + (size_t)(rt * 16 + mrow) * 32 + koff;
        pa[rt] = cat8(*(const h8v*)p, *(const h8v*)(p + 16));
      }
    }

    // ---- O += P.V, V fragments shared across row subtiles ----
#pragma unroll
    for (int t = 0; t < 4; ++t) {
      h16 vb = load_b16(Vh + (size_t)(t * 16) * TT + s0, TT);
      o[0][t] = WMMA_F16(pa[0], vb, o[0][t]);
      o[1][t] = WMMA_F16(pa[1], vb, o[1][t]);
    }
    __syncthreads();  // LDS reused next iteration
  }

  // ---- normalize and write f16 AttnOut [B,T,C] at column h*HD ----
  _Float16* Ob = Out + ((size_t)b * TT + t0) * CC + h * HD;
#pragma unroll
  for (int rt = 0; rt < 2; ++rt)
#pragma unroll
    for (int r = 0; r < 8; ++r) {
      const float inv = 1.0f / l[rt][r];
      const int M = rt * 16 + r + 8 * hi;
#pragma unroll
      for (int t = 0; t < 4; ++t)
        Ob[(size_t)M * CC + t * 16 + nn] = (_Float16)(o[rt][t][r] * inv);
    }
}

// ============================================================
// Kernel 3: output projection y = x @ Wp^T + bp, 32x64 tile/wave.
// grid = (B*T/32, C/64), block = 32. Wp16 is [n][k] natively.
// ============================================================
__global__ void out_proj_kernel(const _Float16* __restrict__ Xin,
                                const _Float16* __restrict__ Wp16,
                                const float* __restrict__ bp,
                                float* __restrict__ Y) {
  const int n0   = blockIdx.y * 64;
  const int row0 = blockIdx.x * 32;
  const _Float16* Xr = Xin + (size_t)row0 * CC;

  f8 acc[2][4] = {};
  for (int k0 = 0; k0 < CC; k0 += 32) {
    if (k0 + 32 < CC) __builtin_prefetch(Xr + k0 + 32, 0, 0);
    h16 a0 = load_a16(Xr + k0, CC);
    h16 a1 = load_a16(Xr + (size_t)16 * CC + k0, CC);
#pragma unroll
    for (int t = 0; t < 4; ++t) {
      h16 bm = load_b16(Wp16 + (size_t)(n0 + t * 16) * CC + k0, CC);
      acc[0][t] = WMMA_F16(a0, bm, acc[0][t]);
      acc[1][t] = WMMA_F16(a1, bm, acc[1][t]);
    }
  }

  const int lane = threadIdx.x & 31;
  const int nn   = lane & 15;
  const int hi   = lane >> 4;
#pragma unroll
  for (int rt = 0; rt < 2; ++rt)
#pragma unroll
    for (int t = 0; t < 4; ++t)
#pragma unroll
      for (int r = 0; r < 8; ++r) {
        const int M   = row0 + rt * 16 + r + 8 * hi;
        const int col = n0 + t * 16 + nn;
        Y[(size_t)M * CC + col] = acc[rt][t][r] + bp[col];
      }
}

// ============================================================
// Host-side launch
// ============================================================
extern "C" void kernel_launch(void* const* d_in, const int* in_sizes, int n_in,
                              void* d_out, int out_size, void* d_ws, size_t ws_size,
                              hipStream_t stream) {
  (void)in_sizes; (void)n_in; (void)out_size; (void)ws_size;

  const float* q  = (const float*)d_in[0];
  const float* k  = (const float*)d_in[1];
  const float* v  = (const float*)d_in[2];
  const float* Wq = (const float*)d_in[3];
  const float* Wk = (const float*)d_in[4];
  const float* Wv = (const float*)d_in[5];
  const float* Wp = (const float*)d_in[6];
  const float* bp = (const float*)d_in[7];
  float* out = (float*)d_out;

  const size_t NX = (size_t)BN * TT * CC;   // 3,145,728 elems
  const size_t NW = (size_t)HH * CC * HD;   //   589,824 elems

  _Float16* X16q = (_Float16*)d_ws;
  _Float16* X16k = X16q + NX;
  _Float16* X16v = X16k + NX;
  _Float16* Q16  = X16v + NX;               // [B,H,T,HD]
  _Float16* K16  = Q16 + NX;                // [B,H,T,HD]
  _Float16* Vt16 = K16 + NX;                // [B,H,HD,T]
  _Float16* AO16 = Vt16 + NX;               // [B,T,C]
  _Float16* Wqt  = AO16 + NX;               // [H,HD,C]
  _Float16* Wkt  = Wqt + NW;
  _Float16* Wvt  = Wkt + NW;
  _Float16* Wp16 = Wvt + NW;                // [C,C] (used as [n][k])

  // --- one-time f16 materialization ---
  const int nx8 = (int)(NX / 8);
  cvt_f16_kernel<<<(nx8 + 255) / 256, 256, 0, stream>>>(q, X16q, nx8);
  cvt_f16_kernel<<<(nx8 + 255) / 256, 256, 0, stream>>>(k, X16k, nx8);
  cvt_f16_kernel<<<(nx8 + 255) / 256, 256, 0, stream>>>(v, X16v, nx8);
  const int nw8 = (CC * CC) / 8;
  cvt_f16_kernel<<<(nw8 + 255) / 256, 256, 0, stream>>>(Wp, Wp16, nw8);
  {
    const int nth = HH * HD * (CC / 8);
    dim3 g((nth + 255) / 256, 1, 3);
    wqkv_t_kernel<<<g, 256, 0, stream>>>(Wq, Wk, Wv, Wqt, Wkt, Wvt);
  }

  // --- QKV projection ---
  dim3 g1(BN * (TT / 32), HH, 3);
  qkv_gemm_kernel<<<g1, 32, 0, stream>>>(X16q, X16k, X16v, Wqt, Wkt, Wvt,
                                         Q16, K16, Vt16);

  // --- causal flash attention ---
  dim3 g2(TT / 32, HH, BN);
  attn_kernel<<<g2, 32, 0, stream>>>(Q16, K16, Vt16, AO16);

  // --- output projection ---
  dim3 g3(BN * TT / 32, CC / 64, 1);
  out_proj_kernel<<<g3, 32, 0, stream>>>(AO16, Wp16, bp, out);
}
